// CompressiveMemory_20753281974428
// MI455X (gfx1250) — matrix operand order repaired and checked by reading, not compiled, not run on hardware
//
#include <hip/hip_runtime.h>
#include <hip/hip_bf16.h>

// ---------------------------------------------------------------------------
// Compressive-memory attention (Infini-attention, delta rule) for gfx1250.
// Heavy matmuls -> v_wmma_f32_16x16x32_bf16 (f32 accum). B-operands staged
// through LDS (ds_load_b128 fragments); async global->LDS staging when the
// toolchain exposes the gfx1250 async builtins (ASYNCcnt-tracked).
// ---------------------------------------------------------------------------

#define BB   2
#define SS   4096
#define DD   2048
#define HH   16
#define DKK  128
#define DVV  128
#define SEGL 512
#define NSEG 8
#define MS   (BB * SS)          // 8192 rows
#define FD   (HH * DKK)         // 2048 feature dim

#if __has_builtin(__builtin_amdgcn_global_load_async_to_lds_b128) && \
    __has_builtin(__builtin_amdgcn_s_wait_asynccnt)
#define HAVE_ASYNC_LDS 1
typedef int vi4 __attribute__((vector_size(16)));
typedef __attribute__((address_space(1))) vi4* gptr_vi4;   // global (AS1)
typedef __attribute__((address_space(3))) vi4* lptr_vi4;   // LDS (AS3)
#endif

typedef unsigned short u16;
typedef __attribute__((ext_vector_type(16))) __bf16 v16bf;
typedef __attribute__((ext_vector_type(8)))  float  v8f;

union Frag16 {
    v16bf bf;
    u16   u[16];
    uint4 q[2];
};
union Pack8 {
    uint4 q;
    u16   u[8];
};

__device__ __forceinline__ u16 f2bf(float f) {
    union { float f; unsigned u; } c; c.f = f;
    if ((c.u & 0x7fffffffu) > 0x7f800000u) return (u16)0x7fc0;   // NaN
    unsigned r = c.u + 0x7fffu + ((c.u >> 16) & 1u);             // RNE
    return (u16)(r >> 16);
}
__device__ __forceinline__ float bf2f(u16 h) {
    union { unsigned u; float f; } c; c.u = ((unsigned)h) << 16;
    return c.f;
}
__device__ __forceinline__ float sigma_elu(float x) {            // elu(x)+1
    return x > 0.f ? x + 1.f : __expf(x);
}
__device__ __forceinline__ v8f wmma_bf(const Frag16& a, const Frag16& b, v8f c) {
    return __builtin_amdgcn_wmma_f32_16x16x32_bf16(
        false, a.bf, false, b.bf, (short)0, c, false, false);
}

// ---------------------------------------------------------------------------
// elementwise f32 -> bf16
// ---------------------------------------------------------------------------
__global__ void cvt_f32_bf16(const float* __restrict__ in, u16* __restrict__ out, int n) {
    int i = blockIdx.x * blockDim.x + threadIdx.x;
    if (i < n) out[i] = f2bf(in[i]);
}

// ---------------------------------------------------------------------------
// C(MxN) = A(MxK,bf16,rowmajor) x B(KxN,bf16,rowmajor). grid=(N/16, M/128),
// block=256 (8 waves); each wave owns one 16x16 C tile; the 32x16 B tile is
// staged once per block in LDS (async-to-LDS when the toolchain has it).
// ---------------------------------------------------------------------------
template <bool OUT_BF16>
__global__ void gemm_bf16_wmma(const u16* __restrict__ A, const u16* __restrict__ B,
                               void* __restrict__ C, int M, int N, int K) {
    __shared__ u16 bTile[512];                 // 1 KiB staging for 32x16 B tile
    const int tid  = threadIdx.x;
    const int wave = tid >> 5;
    const int lane = tid & 31;
    const int half = lane >> 4;
    const int r    = lane & 15;
    const int m0 = blockIdx.y * 128 + wave * 16;
    const int n0 = blockIdx.x * 16;
    const u16* arow = A + (size_t)(m0 + r) * K;
    v8f acc = {};
    for (int k0 = 0; k0 < K; k0 += 32) {
#if defined(HAVE_ASYNC_LDS)
        // async copy: bTile row-major [k][n] (32x16), 64 x 16B chunks
        if (tid < 64) {
            const int row = tid >> 1, c = tid & 1;
            __builtin_amdgcn_global_load_async_to_lds_b128(
                (gptr_vi4)(B + (size_t)(k0 + row) * N + n0 + c * 8),
                (lptr_vi4)(bTile + row * 16 + c * 8),
                0, 0);
        }
        __builtin_amdgcn_s_wait_asynccnt(0);
        __syncthreads();
#else
        // manual transposed staging: bTile [n][k] (16x32)
        if (tid < 64) {
            const int row = tid >> 1, c = tid & 1;
            Pack8 w;
            w.q = *(const uint4*)(B + (size_t)(k0 + row) * N + n0 + c * 8);
#pragma unroll
            for (int i = 0; i < 8; ++i) bTile[(c * 8 + i) * 32 + row] = w.u[i];
        }
        __syncthreads();
#endif
        Frag16 a, b;
        const u16* ap = arow + k0 + half * 16;
        a.q[0] = *(const uint4*)(ap);
        a.q[1] = *(const uint4*)(ap + 8);
        __builtin_prefetch(ap + 64, 0, 1);            // global_prefetch two k-tiles ahead
#if defined(HAVE_ASYNC_LDS)
#pragma unroll
        for (int e = 0; e < 16; ++e) b.u[e] = bTile[(half * 16 + e) * 16 + r];
#else
        const u16* bp = bTile + r * 32 + half * 16;
        b.q[0] = *(const uint4*)(bp);                 // ds_load_b128
        b.q[1] = *(const uint4*)(bp + 8);
#endif
        acc = wmma_bf(a, b, acc);
        __syncthreads();
    }
#pragma unroll
    for (int j = 0; j < 8; ++j) {
        size_t idx = (size_t)(m0 + half * 8 + j) * N + n0 + r;
        if (OUT_BF16) ((u16*)C)[idx] = f2bf(acc[j]);
        else          ((float*)C)[idx] = acc[j];
    }
}

// ---------------------------------------------------------------------------
// In-segment causal softmax attention. One wave per (b,h,seg,16-row strip).
// dyn LDS 48 KiB: [0,32K) f32 scores (later overlaid by transposed V slab),
// [32K,48K) bf16 probabilities. attdot layout: (B,H,S,DV) f32.
// ---------------------------------------------------------------------------
__global__ void attn_kernel(const u16* __restrict__ Qb, const u16* __restrict__ Kb,
                            const u16* __restrict__ Vb, float* __restrict__ attdot) {
    extern __shared__ char smem[];
    float* sc = (float*)smem;                       // [16][512] f32
    u16*   pb = (u16*)(smem + 16 * 512 * 4);        // [16][512] bf16
    u16*   vT = (u16*)smem;                         // [128][32] bf16 (overlays sc)

    int bid = blockIdx.x;
    const int rt  = bid & 31; bid >>= 5;
    const int seg = bid & 7;  bid >>= 3;
    const int h   = bid & 15; bid >>= 4;
    const int b   = bid;

    const int lane = threadIdx.x;
    const int half = lane >> 4;
    const int r    = lane & 15;
    const int s0   = seg * SEGL + rt * 16;          // strip base within sequence
    const size_t segrow = (size_t)b * SS + (size_t)seg * SEGL;

    // ---- scores = Q Kt / sqrt(dk), causal mask -----------------------------
    const u16* qrow = Qb + ((size_t)b * SS + s0 + r) * FD + h * DKK;
    for (int ct = 0; ct < 32; ++ct) {
        const u16* krow = Kb + (segrow + ct * 16 + r) * FD + h * DKK;
        v8f acc = {};
        for (int k0 = 0; k0 < DKK; k0 += 32) {
            Frag16 a, bb;
            const u16* ap = qrow + k0 + half * 16;
            a.q[0] = *(const uint4*)(ap);
            a.q[1] = *(const uint4*)(ap + 8);
            const u16* bp = krow + k0 + half * 16;   // contiguous K per lane
            bb.q[0] = *(const uint4*)(bp);
            bb.q[1] = *(const uint4*)(bp + 8);
            acc = wmma_bf(a, bb, acc);
        }
#pragma unroll
        for (int j = 0; j < 8; ++j) {
            int srow = rt * 16 + half * 8 + j;       // row within segment
            int tcol = ct * 16 + r;                  // col within segment
            float v = acc[j] * 0.08838834764831845f; // 1/sqrt(128)
            if (tcol > srow) v = -__builtin_inff();
            sc[(half * 8 + j) * 512 + tcol] = v;
        }
    }
    __syncthreads();

    // ---- row softmax (lanes 0..15, one row each) ---------------------------
    if (lane < 16) {
        float* row = sc + lane * 512;
        float m = -__builtin_inff();
        for (int t = 0; t < 512; ++t) m = fmaxf(m, row[t]);
        float ssum = 0.f;
        for (int t = 0; t < 512; ++t) { float e = __expf(row[t] - m); ssum += e; row[t] = e; }
        float inv = 1.f / ssum;
        for (int t = 0; t < 512; ++t) pb[lane * 512 + t] = f2bf(row[t] * inv);
    }
    __syncthreads();

    // ---- O = P @ V, V slabs staged transposed in LDS ----------------------
    v8f acc[8] = {};
    for (int k0 = 0; k0 < SEGL; k0 += 32) {
        {   // stage V[k0..k0+31][0..127] -> vT[n][kloc]; lane owns one row
            const u16* vrow = Vb + (segrow + k0 + lane) * FD + h * DVV;
#pragma unroll
            for (int c = 0; c < 16; ++c) {
                Pack8 w;
                w.q = *(const uint4*)(vrow + c * 8);
#pragma unroll
                for (int i = 0; i < 8; ++i) vT[(c * 8 + i) * 32 + lane] = w.u[i];
            }
        }
        __syncthreads();
        Frag16 a;
        const u16* pp = pb + r * 512 + k0 + half * 16;    // ds_load_b128
        a.q[0] = *(const uint4*)(pp);
        a.q[1] = *(const uint4*)(pp + 8);
#pragma unroll
        for (int nt = 0; nt < 8; ++nt) {
            Frag16 bb;
            const u16* bp = vT + (nt * 16 + r) * 32 + half * 16;
            bb.q[0] = *(const uint4*)(bp);                // ds_load_b128
            bb.q[1] = *(const uint4*)(bp + 8);
            acc[nt] = wmma_bf(a, bb, acc[nt]);
        }
        __syncthreads();
    }
#pragma unroll
    for (int nt = 0; nt < 8; ++nt)
#pragma unroll
        for (int j = 0; j < 8; ++j) {
            size_t idx = (((size_t)b * HH + h) * SS + s0 + half * 8 + j) * DVV + nt * 16 + r;
            attdot[idx] = acc[nt][j];
        }
}

// ---------------------------------------------------------------------------
// Sequential compressive-memory recurrence. One 256-thread block per (b,h).
// LDS-resident state (dyn LDS ~100.5 KiB): f32 memory, transposed bf16 shadow
// [n][k], z, kn, qn. sigk / sigkT / deltaT live in global scratch.
// attb is written in (B, NSEG, H, SEGL, DV) order == the reference's
// faithful-to-torch flat (B, S, H*DV) view of contiguous (B,H,SEG,DV).
// ---------------------------------------------------------------------------
__global__ void memory_kernel(const u16* __restrict__ Qb, const u16* __restrict__ Kb,
                              const u16* __restrict__ Vb, const float* __restrict__ attdot,
                              const float* __restrict__ betas,
                              u16* __restrict__ sigk, u16* __restrict__ sigkT,
                              u16* __restrict__ deltaT, u16* __restrict__ attb) {
    extern __shared__ char smem[];
    float* memF  = (float*)smem;                     // [128][128] f32, 64 KiB
    u16*   membT = (u16*)(smem + 65536);             // [n][k] bf16, 32 KiB
    float* zS    = (float*)(smem + 98304);           // [128]
    float* knS   = zS + DKK;                         // [512]
    float* qnS   = knS + SEGL;                       // [512]

    const int bh = blockIdx.x;            // b*H + h
    const int b  = bh >> 4;
    const int h  = bh & 15;
    const int tid  = threadIdx.x;
    const int wave = tid >> 5;
    const int lane = tid & 31;
    const int half = lane >> 4;
    const int r    = lane & 15;

    u16* sigkP  = sigk  + (size_t)bh * SEGL * DKK;   // [s][d]
    u16* sigkTP = sigkT + (size_t)bh * DKK * SEGL;   // [d][s]
    u16* deltTP = deltaT + (size_t)bh * DVV * SEGL;  // [n][s]

    for (int i = tid; i < DKK * DVV; i += 256) { memF[i] = 0.f; membT[i] = 0; }
    for (int i = tid; i < DKK; i += 256) zS[i] = 1.0f / 128.0f;
    __syncthreads();

    for (int seg = 0; seg < NSEG; ++seg) {
        const size_t segrow = (size_t)b * SS + (size_t)seg * SEGL;

        // phase 1: sigma_k = elu(k)+1, bf16 (row-major + transposed)
        for (int i = tid; i < SEGL * DKK; i += 256) {
            int s = i >> 7, d = i & 127;
            float kv = bf2f(Kb[(segrow + s) * FD + h * DKK + d]);
            u16 sv = f2bf(sigma_elu(kv));
            sigkP[i] = sv;
            sigkTP[d * SEGL + s] = sv;
        }
        __threadfence(); __syncthreads();

        // phase 2: kn = sigk.z , qn = sigq.z (pre-update z)
        for (int s = tid; s < SEGL; s += 256) {
            const u16* qrow = Qb + (segrow + s) * FD + h * DKK;
            float aK = 0.f, aQ = 0.f;
            for (int d = 0; d < DKK; ++d) {
                float zd = zS[d];
                aK += bf2f(sigkP[s * DKK + d]) * zd;
                aQ += sigma_elu(bf2f(qrow[d])) * zd;
            }
            knS[s] = aK; qnS[s] = aQ;
        }
        __syncthreads();

        // phase 3: retrieval (pre-update mem), deltaT, beta-blend output
        for (int rt4 = 0; rt4 < 4; ++rt4) {
            const int sloc0 = (wave * 4 + rt4) * 16;
            for (int nt = 0; nt < 8; ++nt) {
                v8f accQ = {}, accK = {};
                for (int k0 = 0; k0 < DKK; k0 += 32) {
                    Frag16 aq, ak, bbm;
                    const u16* qrow = Qb + (segrow + sloc0 + r) * FD + h * DKK + k0 + half * 16;
#pragma unroll
                    for (int i = 0; i < 8; ++i) {
                        aq.u[i]     = f2bf(sigma_elu(bf2f(qrow[i])));
                        aq.u[i + 8] = f2bf(sigma_elu(bf2f(qrow[i + 8])));
                    }
                    const u16* kr = sigkP + (sloc0 + r) * DKK + k0 + half * 16;
                    ak.q[0] = *(const uint4*)(kr);
                    ak.q[1] = *(const uint4*)(kr + 8);
                    const u16* bp = membT + (nt * 16 + r) * DKK + k0 + half * 16;
                    bbm.q[0] = *(const uint4*)(bp);       // ds_load_b128
                    bbm.q[1] = *(const uint4*)(bp + 8);
                    accQ = wmma_bf(aq, bbm, accQ);
                    accK = wmma_bf(ak, bbm, accK);
                }
#pragma unroll
                for (int j = 0; j < 8; ++j) {
                    const int sloc = sloc0 + half * 8 + j;
                    const int n    = nt * 16 + r;
                    float attmem = accQ[j] / qnS[sloc];
                    float kret   = accK[j] / knS[sloc];
                    float vv = bf2f(Vb[(segrow + sloc) * FD + h * DVV + n]);
                    deltTP[n * SEGL + sloc] = f2bf(vv - kret);
                    float bsig = 1.f / (1.f + __expf(-betas[h * DVV + n]));
                    float ad   = attdot[((size_t)bh * SS + seg * SEGL + sloc) * DVV + n];
                    float att  = bsig * attmem + (1.f - bsig) * ad;
                    attb[((((size_t)b * NSEG + seg) * HH + h) * SEGL + sloc) * DVV + n] = f2bf(att);
                }
            }
        }
        __threadfence(); __syncthreads();

        // phase 4: mem += sigk^T @ delta ; z += col-sum(sigk)
        for (int t8 = 0; t8 < 8; ++t8) {
            const int tile = wave * 8 + t8;
            const int mt = tile >> 3, nt = tile & 7;
            v8f acc = {};
            for (int k0 = 0; k0 < SEGL; k0 += 32) {
                Frag16 a, bb;
                const u16* ar = sigkTP + (size_t)(mt * 16 + r) * SEGL + k0 + half * 16;
                a.q[0] = *(const uint4*)(ar);             // contiguous (transposed store)
                a.q[1] = *(const uint4*)(ar + 8);
                const u16* br = deltTP + (size_t)(nt * 16 + r) * SEGL + k0 + half * 16;
                bb.q[0] = *(const uint4*)(br);
                bb.q[1] = *(const uint4*)(br + 8);
                acc = wmma_bf(a, bb, acc);
            }
#pragma unroll
            for (int j = 0; j < 8; ++j) {
                int m = mt * 16 + half * 8 + j, n = nt * 16 + r;
                float nv = memF[m * DVV + n] + acc[j];
                memF[m * DVV + n]  = nv;
                membT[n * DKK + m] = f2bf(nv);
            }
        }
        for (int d = tid; d < DKK; d += 256) {
            float acc = 0.f;
            const u16* srow = sigkTP + (size_t)d * SEGL;
            for (int s = 0; s < SEGL; ++s) acc += bf2f(srow[s]);
            zS[d] += acc;
        }
        __syncthreads();
    }
}

// ---------------------------------------------------------------------------
extern "C" void kernel_launch(void* const* d_in, const int* in_sizes, int n_in,
                              void* d_out, int out_size, void* d_ws, size_t ws_size,
                              hipStream_t stream) {
    const float* x     = (const float*)d_in[0];
    const float* Wq    = (const float*)d_in[1];
    const float* Wk    = (const float*)d_in[2];
    const float* Wv    = (const float*)d_in[3];
    const float* Wo    = (const float*)d_in[4];
    const float* betas = (const float*)d_in[5];

    char* ws = (char*)d_ws;
    size_t off = 0;
    auto alloc = [&](size_t bytes) -> void* {
        off = (off + 255) & ~(size_t)255;
        void* p = ws + off;
        off += bytes;
        return p;
    };
    u16*   xb     = (u16*)alloc((size_t)MS * DD * 2);
    u16*   wqb    = (u16*)alloc((size_t)DD * FD * 2);
    u16*   wkb    = (u16*)alloc((size_t)DD * FD * 2);
    u16*   wvb    = (u16*)alloc((size_t)DD * FD * 2);
    u16*   wob    = (u16*)alloc((size_t)FD * DD * 2);
    u16*   qb     = (u16*)alloc((size_t)MS * FD * 2);
    u16*   kb     = (u16*)alloc((size_t)MS * FD * 2);
    u16*   vb     = (u16*)alloc((size_t)MS * FD * 2);
    float* attdot = (float*)alloc((size_t)BB * HH * SS * DVV * 4);
    u16*   attb   = (u16*)alloc((size_t)MS * FD * 2);
    u16*   sigkb  = (u16*)alloc((size_t)BB * HH * SEGL * DKK * 2);
    u16*   sigkTb = (u16*)alloc((size_t)BB * HH * DKK * SEGL * 2);
    u16*   deltTb = (u16*)alloc((size_t)BB * HH * DVV * SEGL * 2);

    // 1) casts to bf16
    int nx = MS * DD;
    cvt_f32_bf16<<<(nx + 255) / 256, 256, 0, stream>>>(x, xb, nx);
    int nw = DD * FD;
    cvt_f32_bf16<<<(nw + 255) / 256, 256, 0, stream>>>(Wq, wqb, nw);
    cvt_f32_bf16<<<(nw + 255) / 256, 256, 0, stream>>>(Wk, wkb, nw);
    cvt_f32_bf16<<<(nw + 255) / 256, 256, 0, stream>>>(Wv, wvb, nw);
    cvt_f32_bf16<<<(nw + 255) / 256, 256, 0, stream>>>(Wo, wob, nw);

    // 2) projections (WMMA GEMM, bf16 out)
    dim3 gP(FD / 16, MS / 128);
    gemm_bf16_wmma<true><<<gP, 256, 0, stream>>>(xb, wqb, qb, MS, FD, DD);
    gemm_bf16_wmma<true><<<gP, 256, 0, stream>>>(xb, wkb, kb, MS, FD, DD);
    gemm_bf16_wmma<true><<<gP, 256, 0, stream>>>(xb, wvb, vb, MS, FD, DD);

    // 3) per-segment causal softmax attention (WMMA + LDS softmax)
    attn_kernel<<<BB * HH * NSEG * (SEGL / 16), 32, 48 * 1024, stream>>>(qb, kb, vb, attdot);

    // 4) sequential compressive-memory recurrence (32 chains, LDS-resident state)
    memory_kernel<<<BB * HH, 256, 102912, stream>>>(qb, kb, vb, attdot, betas,
                                                    sigkb, sigkTb, deltTb, attb);

    // 5) output projection (WMMA GEMM, f32 out)
    dim3 gO(DD / 16, MS / 128);
    gemm_bf16_wmma<false><<<gO, 256, 0, stream>>>(attb, wob, d_out, MS, DD, FD);
}
